// BaselineEncoderDecoderModel_47699906789832
// MI455X (gfx1250) — matrix operand order repaired and checked by reading, not compiled, not run on hardware
//
#include <hip/hip_runtime.h>
#include <hip/hip_bf16.h>
#include <math.h>

typedef __attribute__((ext_vector_type(16))) _Float16 v16h;
typedef __attribute__((ext_vector_type(8)))  _Float16 v8h;
typedef __attribute__((ext_vector_type(8)))  float    v8f;

#define HD     500      // hidden size
#define H3     1500     // 3*H
#define KPAD   512      // padded K (multiple of 32)
#define NPADIH 1504     // 1500 padded to multiple of 16
#define SEQ    512
#define VOCAB  50000    // 3125 N-tiles exactly
#define VTILES 3125
#define VCHUNKS 125     // V-chunks for fused softmax partials
#define VT_PER_CHUNK 25 // 3125 / 125

#define GRU_BLOCKS  12
#define GRU_THREADS 128

// ---------------- f32 -> f16 conversion with row/K padding ----------------
__global__ void cvt_pad_f16_kernel(const float* __restrict__ src, _Float16* __restrict__ dst,
                                   int srcRows, int srcK, int dstRows, int dstK) {
    int idx = blockIdx.x * blockDim.x + threadIdx.x;
    int total = dstRows * dstK;
    if (idx >= total) return;
    int r = idx / dstK, k = idx % dstK;
    float v = (r < srcRows && k < srcK) ? src[(size_t)r * srcK + k] : 0.0f;
    dst[idx] = (_Float16)v;
}

// ---------------- embedding gathers (f16, K-padded) ----------------
__global__ void embed_enc_kernel(const int* __restrict__ tok, const float* __restrict__ emb,
                                 _Float16* __restrict__ x) {
    int idx = blockIdx.x * blockDim.x + threadIdx.x;
    if (idx >= SEQ * KPAD) return;
    int t = idx / KPAD, k = idx % KPAD;
    float v = (k < HD) ? emb[(size_t)tok[t] * HD + k] : 0.0f;
    x[idx] = (_Float16)v;
}

__global__ void embed_dec_kernel(const int* __restrict__ tgt, const float* __restrict__ emb,
                                 _Float16* __restrict__ x) {
    int idx = blockIdx.x * blockDim.x + threadIdx.x;
    if (idx >= SEQ * KPAD) return;
    int t = idx / KPAD, k = idx % KPAD;
    float v = 0.0f;
    if (k < HD) {
        int token = (t == 0) ? 0 : tgt[t - 1];          // teacher forcing, PAD first
        v = emb[(size_t)token * HD + k];
        v = v > 0.0f ? v : 0.0f;                        // relu
    }
    x[idx] = (_Float16)v;
}

// ---------------- WMMA GEMM: C[M x Cstride] = A[M x K] * W[N x K]^T + bias ----------------
__global__ void wmma_gemm_nt_kernel(const _Float16* __restrict__ A,
                                    const _Float16* __restrict__ W,
                                    const float* __restrict__ bias, int biasN,
                                    float* __restrict__ C, int Cstride,
                                    int M, int N, int K) {
    int wave = (blockIdx.x * blockDim.x + threadIdx.x) >> 5;
    int ntN = N >> 4;
    int total = (M >> 4) * ntN;
    if (wave >= total) return;                  // wave-uniform, EXEC stays all-ones
    int mt = wave / ntN, nt = wave % ntN;
    int lane = threadIdx.x & 31;
    int hi = lane >> 4, lo = lane & 15;
    const _Float16* Ab = A + (size_t)(mt * 16 + lo) * K;   // A frag: lane = row M
    const _Float16* Wb = W + (size_t)(nt * 16 + lo) * K;   // B frag: lane = col N
    v8f acc = {};
    for (int k0 = 0; k0 < K; k0 += 32) {
        union { v16h v; v8h h[2]; } a;
        a.v = *reinterpret_cast<const v16h*>(Ab + k0 + hi * 16);
        v16h b = *reinterpret_cast<const v16h*>(Wb + k0 + hi * 16);
        acc = __builtin_amdgcn_wmma_f32_16x16x32_f16(false, a.v, false, b,
                                                     (short)0, acc, false, false);
    }
    int col = nt * 16 + lo;
    float bv = (bias != nullptr && col < biasN) ? bias[col] : 0.0f;
    float* Cr = C + (size_t)(mt * 16 + hi * 8) * Cstride + col;
#pragma unroll
    for (int i = 0; i < 8; i++) Cr[(size_t)i * Cstride] = acc[i] + bv;
}

// ---------------- device-scope sense barrier (co-resident small grid) ----------------
__device__ __forceinline__ void grid_barrier(int* cnt, int* gen, int nb) {
    __threadfence();
    __syncthreads();
    if (threadIdx.x == 0) {
        int g = __hip_atomic_load(gen, __ATOMIC_ACQUIRE, __HIP_MEMORY_SCOPE_AGENT);
        int a = __hip_atomic_fetch_add(cnt, 1, __ATOMIC_ACQ_REL, __HIP_MEMORY_SCOPE_AGENT);
        if (a == nb - 1) {
            __hip_atomic_store(cnt, 0, __ATOMIC_RELAXED, __HIP_MEMORY_SCOPE_AGENT);
            __hip_atomic_fetch_add(gen, 1, __ATOMIC_ACQ_REL, __HIP_MEMORY_SCOPE_AGENT);
        } else {
            while (__hip_atomic_load(gen, __ATOMIC_ACQUIRE, __HIP_MEMORY_SCOPE_AGENT) == g)
                __builtin_amdgcn_s_sleep(1);
        }
    }
    __syncthreads();
}

__global__ void scan_init_kernel(const float* __restrict__ h0, float* __restrict__ hbuf,
                                 int* __restrict__ bar) {
    int j = threadIdx.x;                 // 512
    if (j < HD) hbuf[j] = h0[j];
    if (j < 2)  bar[j] = 0;
}

__device__ __forceinline__ float sigm(float x) { return 1.0f / (1.0f + expf(-x)); }

// ---------------- multi-WGP sequential GRU scan ----------------
// Per step: 12 blocks split the 1500 gh rows (f16 Whh, f32 acc), barrier,
// split the gate/update rows, barrier. h ping-pongs through a 2KB L2-hot buffer.
__global__ void gru_scan_mb_kernel(const float* __restrict__ G, int Gstride,
                                   const _Float16* __restrict__ Whh16,  // H3 x KPAD f16
                                   const float* __restrict__ bhh,
                                   float* __restrict__ hbuf0, float* __restrict__ hbuf1,
                                   float* __restrict__ ghbuf,
                                   int* __restrict__ bar,
                                   float* __restrict__ hlast,
                                   _Float16* __restrict__ hs,           // may be null; S x KPAD
                                   int S) {
    __shared__ float h_lds[KPAD];
    const int tid = threadIdx.x, blk = blockIdx.x, nb = gridDim.x;
    float* hcur = hbuf0; float* hnext = hbuf1;
    const int rpb = (H3 + nb - 1) / nb;          // 125 gh rows per block
    const int cpb = (KPAD + nb - 1) / nb;        // 43 update cols per block

    for (int t = 0; t < S; t++) {
        for (int j = tid; j < KPAD; j += blockDim.x)
            h_lds[j] = (j < HD) ? hcur[j] : 0.0f;
        __syncthreads();

        // phase 1: gh = bhh + Whh * h
        int r0 = blk * rpb, r1 = (r0 + rpb < H3) ? r0 + rpb : H3;
        for (int j = r0 + tid; j < r1; j += blockDim.x) {
            const _Float16* w = Whh16 + (size_t)j * KPAD;
            float acc = bhh[j];
            for (int k = 0; k < KPAD; k += 8) {
                v8h wv = *reinterpret_cast<const v8h*>(w + k);
#pragma unroll
                for (int u = 0; u < 8; u++) acc = fmaf((float)wv[u], h_lds[k + u], acc);
            }
            ghbuf[j] = acc;
        }
        grid_barrier(&bar[0], &bar[1], nb);

        // phase 2: gates + hidden update
        const float* g = G + (size_t)t * Gstride;
        int c0 = blk * cpb, c1 = (c0 + cpb < KPAD) ? c0 + cpb : KPAD;
        for (int j = c0 + tid; j < c1; j += blockDim.x) {
            if (j < HD) {
                float r  = sigm(g[j]          + ghbuf[j]);
                float z  = sigm(g[HD + j]     + ghbuf[HD + j]);
                float n  = tanhf(g[2 * HD + j] + r * ghbuf[2 * HD + j]);
                float hv = (1.0f - z) * n + z * h_lds[j];
                hnext[j] = hv;
                if (hs) hs[(size_t)t * KPAD + j] = (_Float16)hv;
            } else if (hs) {
                hs[(size_t)t * KPAD + j] = (_Float16)0.0f;
            }
        }
        grid_barrier(&bar[0], &bar[1], nb);
        float* tmp = hcur; hcur = hnext; hnext = tmp;
    }
    if (blk == 0)
        for (int j = tid; j < HD; j += blockDim.x) hlast[j] = hcur[j];
}

// ---------------- fused vocab GEMM + online log-softmax partials ----------------
// grid = (VCHUNKS, 32 M-tiles), block = 256 (8 waves). Each wave sweeps its
// N-tiles of the chunk, WMMA-accumulates over K, then updates 8 per-row online
// (max, sum, argmax) states in registers. Logits are never written to memory.
__global__ void vocab_fused_kernel(const _Float16* __restrict__ A,   // SEQ x KPAD
                                   const _Float16* __restrict__ W,   // VOCAB x KPAD
                                   const float* __restrict__ bias,
                                   float* __restrict__ pm, float* __restrict__ ps,
                                   int* __restrict__ pidx) {
    __shared__ float lm[8][32][8];
    __shared__ float ls[8][32][8];
    __shared__ int   li[8][32][8];
    int chunk = blockIdx.x;
    int mt    = blockIdx.y;
    int wave  = threadIdx.x >> 5;
    int lane  = threadIdx.x & 31;
    int hi = lane >> 4, lo = lane & 15;
    const _Float16* Ab = A + (size_t)(mt * 16 + lo) * KPAD;

    float m[8], s[8]; int mi[8];
#pragma unroll
    for (int i = 0; i < 8; i++) { m[i] = -INFINITY; s[i] = 0.0f; mi[i] = 0; }

    int ntEnd = (chunk + 1) * VT_PER_CHUNK;
    for (int nt = chunk * VT_PER_CHUNK + wave; nt < ntEnd; nt += 8) {  // wave-uniform
        const _Float16* Wb = W + (size_t)(nt * 16 + lo) * KPAD;
        v8f acc = {};
        for (int k0 = 0; k0 < KPAD; k0 += 32) {
            union { v16h v; v8h h[2]; } a;
            a.v = *reinterpret_cast<const v16h*>(Ab + k0 + hi * 16);
            v16h b = *reinterpret_cast<const v16h*>(Wb + k0 + hi * 16);
            acc = __builtin_amdgcn_wmma_f32_16x16x32_f16(false, a.v, false, b,
                                                         (short)0, acc, false, false);
        }
        int col = nt * 16 + lo;
        float bv = bias[col];
#pragma unroll
        for (int i = 0; i < 8; i++) {       // row = mt*16 + hi*8 + i, this lane's column
            float v = acc[i] + bv;
            if (v > m[i]) { s[i] = s[i] * expf(m[i] - v) + 1.0f; m[i] = v; mi[i] = col; }
            else          { s[i] += expf(v - m[i]); }
        }
    }
#pragma unroll
    for (int i = 0; i < 8; i++) { lm[wave][lane][i] = m[i]; ls[wave][lane][i] = s[i]; li[wave][lane][i] = mi[i]; }
    __syncthreads();

    if (threadIdx.x < 16) {                 // thread r reduces 128 states of row r
        int r = threadIdx.x;
        int i = r & 7, lbase = (r >> 3) * 16;
        float M = -INFINITY, S = 0.0f; int I = 0;
        for (int w = 0; w < 8; w++)
            for (int c = 0; c < 16; c++) {
                float mm = lm[w][lbase + c][i], ssv = ls[w][lbase + c][i];
                int   ii = li[w][lbase + c][i];
                if (mm > M || (mm == M && ii < I)) { S = S * expf(M - mm) + ssv; M = mm; I = ii; }
                else                               { S += ssv * expf(mm - M); }
            }
        size_t p = (size_t)(mt * 16 + r) * VCHUNKS + chunk;
        pm[p] = M; ps[p] = S; pidx[p] = I;
    }
}

// exact logit at the target index (512 small dots; matches fused GEMM precision)
__global__ void target_logit_kernel(const _Float16* __restrict__ hs, const _Float16* __restrict__ W,
                                    const float* __restrict__ bias, const int* __restrict__ tgt,
                                    float* __restrict__ tl) {
    int t = blockIdx.x * blockDim.x + threadIdx.x;
    if (t >= SEQ) return;
    int v = tgt[t];
    const _Float16* a = hs + (size_t)t * KPAD;
    const _Float16* w = W  + (size_t)v * KPAD;
    float s = bias[v];
    for (int k = 0; k < KPAD; k += 8) {
        v8h av = *reinterpret_cast<const v8h*>(a + k);
        v8h wv = *reinterpret_cast<const v8h*>(w + k);
#pragma unroll
        for (int u = 0; u < 8; u++) s = fmaf((float)av[u], (float)wv[u], s);
    }
    tl[t] = s;
}

// merge per-chunk softmax partials -> per-row loss + argmax
__global__ void softmax_final_kernel(const float* __restrict__ pm, const float* __restrict__ ps,
                                     const int* __restrict__ pidx, const float* __restrict__ tl,
                                     float* __restrict__ rowls, float* __restrict__ out) {
    int t = threadIdx.x;                    // 512
    float M = -INFINITY, S = 0.0f; int I = 0;
    for (int c = 0; c < VCHUNKS; c++) {
        float mm = pm[(size_t)t * VCHUNKS + c], ssv = ps[(size_t)t * VCHUNKS + c];
        int   ii = pidx[(size_t)t * VCHUNKS + c];
        if (mm > M || (mm == M && ii < I)) { S = S * expf(M - mm) + ssv; M = mm; I = ii; }
        else                               { S += ssv * expf(mm - M); }
    }
    float logp = tl[t] - (M + logf(S));
    rowls[t] = -logp;
    out[1 + HD + t] = (float)I;             // decoder_outputs as float
}

__global__ void finalize_init_kernel(const float* __restrict__ enc_h,
                                     float* __restrict__ out, int out_size) {
    int i = blockIdx.x * blockDim.x + threadIdx.x;
    if (i >= out_size) return;
    float v = 0.0f;
    if (i >= 1 && i <= HD) v = enc_h[i - 1];
    out[i] = v;
}

__global__ void loss_sum_kernel(const float* __restrict__ rowls, float* __restrict__ out) {
    __shared__ float s[SEQ];
    int tid = threadIdx.x;                  // 512, deterministic tree sum
    s[tid] = rowls[tid];
    __syncthreads();
    for (int st = 256; st > 0; st >>= 1) {
        if (tid < st) s[tid] += s[tid + st];
        __syncthreads();
    }
    if (tid == 0) out[0] = s[0];
}

extern "C" void kernel_launch(void* const* d_in, const int* in_sizes, int n_in,
                              void* d_out, int out_size, void* d_ws, size_t ws_size,
                              hipStream_t stream) {
    const int*   enc_in  = (const int*)  d_in[0];
    const int*   tgt     = (const int*)  d_in[1];
    const float* enc_h0  = (const float*)d_in[2];
    const float* enc_emb = (const float*)d_in[3];
    const float* enc_Wih = (const float*)d_in[4];
    const float* enc_Whh = (const float*)d_in[5];
    const float* enc_bih = (const float*)d_in[6];
    const float* enc_bhh = (const float*)d_in[7];
    const float* dec_emb = (const float*)d_in[8];
    const float* dec_Wih = (const float*)d_in[9];
    const float* dec_Whh = (const float*)d_in[10];
    const float* dec_bih = (const float*)d_in[11];
    const float* dec_bhh = (const float*)d_in[12];
    const float* out_W   = (const float*)d_in[13];
    const float* out_b   = (const float*)d_in[14];
    float* out = (float*)d_out;

    char* ws = (char*)d_ws; size_t off = 0;
    auto alloc = [&](size_t bytes) -> void* {
        void* p = ws + off;
        off += (bytes + 255) & ~(size_t)255;
        return p;
    };
    _Float16* xE      = (_Float16*)alloc((size_t)SEQ * KPAD * 2);
    _Float16* xD      = (_Float16*)alloc((size_t)SEQ * KPAD * 2);
    _Float16* WihE    = (_Float16*)alloc((size_t)NPADIH * KPAD * 2);
    _Float16* WihD    = (_Float16*)alloc((size_t)NPADIH * KPAD * 2);
    _Float16* WhhE16  = (_Float16*)alloc((size_t)H3 * KPAD * 2);
    _Float16* WhhD16  = (_Float16*)alloc((size_t)H3 * KPAD * 2);
    _Float16* Wout    = (_Float16*)alloc((size_t)VOCAB * KPAD * 2);
    float*    Gx      = (float*)   alloc((size_t)SEQ * NPADIH * 4);
    float*    Gd      = (float*)   alloc((size_t)SEQ * NPADIH * 4);
    _Float16* hsD     = (_Float16*)alloc((size_t)SEQ * KPAD * 2);
    float*    ehf     = (float*)   alloc((size_t)HD * 4);
    float*    dhf     = (float*)   alloc((size_t)HD * 4);
    float*    hbuf0   = (float*)   alloc((size_t)KPAD * 4);
    float*    hbuf1   = (float*)   alloc((size_t)KPAD * 4);
    float*    ghbuf   = (float*)   alloc((size_t)H3 * 4);
    int*      bar     = (int*)     alloc(2 * sizeof(int));
    float*    pm      = (float*)   alloc((size_t)SEQ * VCHUNKS * 4);
    float*    ps      = (float*)   alloc((size_t)SEQ * VCHUNKS * 4);
    int*      pidx    = (int*)     alloc((size_t)SEQ * VCHUNKS * 4);
    float*    tl      = (float*)   alloc((size_t)SEQ * 4);
    float*    rowls   = (float*)   alloc((size_t)SEQ * 4);

    // --- weight conversions (f32 -> padded f16) ---
    {
        int tot = NPADIH * KPAD;
        cvt_pad_f16_kernel<<<(tot + 255) / 256, 256, 0, stream>>>(enc_Wih, WihE, H3, HD, NPADIH, KPAD);
        cvt_pad_f16_kernel<<<(tot + 255) / 256, 256, 0, stream>>>(dec_Wih, WihD, H3, HD, NPADIH, KPAD);
        int totH = H3 * KPAD;
        cvt_pad_f16_kernel<<<(totH + 255) / 256, 256, 0, stream>>>(enc_Whh, WhhE16, H3, HD, H3, KPAD);
        cvt_pad_f16_kernel<<<(totH + 255) / 256, 256, 0, stream>>>(dec_Whh, WhhD16, H3, HD, H3, KPAD);
        long long totV = (long long)VOCAB * KPAD;
        cvt_pad_f16_kernel<<<(int)((totV + 255) / 256), 256, 0, stream>>>(out_W, Wout, VOCAB, HD, VOCAB, KPAD);
    }

    // --- embeddings ---
    embed_enc_kernel<<<(SEQ * KPAD + 255) / 256, 256, 0, stream>>>(enc_in, enc_emb, xE);
    embed_dec_kernel<<<(SEQ * KPAD + 255) / 256, 256, 0, stream>>>(tgt, dec_emb, xD);

    // --- hoisted input projections (WMMA): G = x @ Wih.T + bih ---
    {
        int tiles = (SEQ >> 4) * (NPADIH >> 4);
        int blocks = (tiles * 32 + 255) / 256;
        wmma_gemm_nt_kernel<<<blocks, 256, 0, stream>>>(xE, WihE, enc_bih, H3, Gx, NPADIH, SEQ, NPADIH, KPAD);
        wmma_gemm_nt_kernel<<<blocks, 256, 0, stream>>>(xD, WihD, dec_bih, H3, Gd, NPADIH, SEQ, NPADIH, KPAD);
    }

    // --- encoder scan (12 cooperating WGPs, per-step grid barrier) ---
    scan_init_kernel<<<1, 512, 0, stream>>>(enc_h0, hbuf0, bar);
    gru_scan_mb_kernel<<<GRU_BLOCKS, GRU_THREADS, 0, stream>>>(
        Gx, NPADIH, WhhE16, enc_bhh, hbuf0, hbuf1, ghbuf, bar, ehf, (_Float16*)nullptr, SEQ);

    // zero d_out + write enc_h into out[1..500]
    finalize_init_kernel<<<(out_size + 255) / 256, 256, 0, stream>>>(ehf, out, out_size);

    // --- decoder scan (h0 = enc_h), stores hidden states as padded f16 ---
    scan_init_kernel<<<1, 512, 0, stream>>>(ehf, hbuf0, bar);
    gru_scan_mb_kernel<<<GRU_BLOCKS, GRU_THREADS, 0, stream>>>(
        Gd, NPADIH, WhhD16, dec_bhh, hbuf0, hbuf1, ghbuf, bar, dhf, hsD, SEQ);

    // --- fused vocab GEMM + online log-softmax (logits never hit memory) ---
    {
        dim3 grid(VCHUNKS, SEQ / 16);
        vocab_fused_kernel<<<grid, 256, 0, stream>>>(hsD, Wout, out_b, pm, ps, pidx);
    }
    target_logit_kernel<<<(SEQ + 127) / 128, 128, 0, stream>>>(hsD, Wout, out_b, tgt, tl);
    softmax_final_kernel<<<1, SEQ, 0, stream>>>(pm, ps, pidx, tl, rowls, out);
    loss_sum_kernel<<<1, SEQ, 0, stream>>>(rowls, out);
}
